// CausalShapedAttention_44135083934204
// MI455X (gfx1250) — matrix operand order repaired and checked by reading, not compiled, not run on hardware
//
#include <hip/hip_runtime.h>
#include <hip/hip_bf16.h>

#define B_SZ 2
#define T_SEQ 2048
#define C_EMB 1024
#define NH 16
#define DH 64
#define QKV_LD (3 * C_EMB)

typedef __attribute__((ext_vector_type(16))) __bf16 bf16x16;
typedef __attribute__((ext_vector_type(8)))  __bf16 bf16x8;
typedef __attribute__((ext_vector_type(8)))  float  f32x8;

union bvec16 { bf16x16 v; bf16x8 h2[2]; __bf16 e[16]; };

__device__ __forceinline__ __bf16 f2bf(float f) {
  unsigned int u = __builtin_bit_cast(unsigned int, f);
  unsigned int r = (u + 0x7FFFu + ((u >> 16) & 1u)) >> 16;
  unsigned short s = (unsigned short)r;
  return __builtin_bit_cast(__bf16, s);
}
__device__ __forceinline__ float bf2f(__bf16 b) {
  unsigned short s = __builtin_bit_cast(unsigned short, b);
  unsigned int u = ((unsigned int)s) << 16;
  return __builtin_bit_cast(float, u);
}

__device__ __forceinline__ f32x8 wmma_bf16(const bvec16& a, const bvec16& b, f32x8 c) {
  return __builtin_amdgcn_wmma_f32_16x16x32_bf16(false, a.v, false, b.v,
                                                 (short)0, c, false, false);
}

// CDNA5 async global->LDS copy (ASYNCcnt-tracked); 16B per lane per issue.
// VDST = per-lane LDS byte offset, VADDR = per-lane 64-bit global address.
__device__ __forceinline__ void async_ld_b128(unsigned lds_off, const void* gaddr) {
  unsigned long long ga = (unsigned long long)gaddr;
  asm volatile("global_load_async_to_lds_b128 %0, %1, off"
               :: "v"(lds_off), "v"(ga) : "memory");
}
__device__ __forceinline__ void wait_async0() {
  asm volatile("s_wait_asynccnt 0x0" ::: "memory");
}

// ---------------------------------------------------------------- convert
__global__ __launch_bounds__(256) void k_f32_to_bf16(const float* __restrict__ in,
                                                     __bf16* __restrict__ out, long n) {
  long i = (long)blockIdx.x * blockDim.x + threadIdx.x;
  long stride = (long)gridDim.x * blockDim.x;
  for (; i < n; i += stride) out[i] = f2bf(in[i]);
}

// ---------------------------------------------------------------- GEMM
// C[M,N] = A[M,K] @ Bmat[N,K]^T + bias, A/Bmat bf16 row-major.
// One wave computes a 32x64 tile: 2 A-frags x 4 B-frags -> 8 WMMAs per K-step.
// Operand k order used here (contiguous, same for A and B):
//   A (16-bit 16x32): lane m = l&15; lane half kbA in {0,16}; elem e -> K = kbA + e (contiguous)
//   B (16-bit 32x16): lane n = l&15; lane half kbB in {0,16}; elem e -> K = kbB + e (contiguous)
__global__ __launch_bounds__(128) void k_gemm_bf16_wmma(
    const __bf16* __restrict__ A, const __bf16* __restrict__ Bm,
    const float* __restrict__ bias,
    float* __restrict__ Cf, __bf16* __restrict__ Cb,
    int M, int N, int K) {
  const int lane = threadIdx.x & 31;
  const int wave = threadIdx.x >> 5;
  const int ntn  = N >> 6;                       // 64-wide N tiles
  const int tid  = blockIdx.x * (blockDim.x >> 5) + wave;
  const int tm   = tid / ntn;
  const int tn   = tid % ntn;
  if (tm >= (M >> 5)) return;  // wave-uniform

  const int nloc = lane & 15;
  const int kbA  = (lane < 16) ? 0 : 16;
  const int kbB  = (lane < 16) ? 0 : 16;

  const __bf16* arow0 = A + (size_t)(tm * 32 + nloc) * K;
  const __bf16* arow1 = arow0 + (size_t)16 * K;
  const __bf16* brow0 = Bm + (size_t)(tn * 64 + nloc) * K;
  const __bf16* brow1 = brow0 + (size_t)16 * K;
  const __bf16* brow2 = brow0 + (size_t)32 * K;
  const __bf16* brow3 = brow0 + (size_t)48 * K;

  f32x8 acc[2][4] = {{{}, {}, {}, {}}, {{}, {}, {}, {}}};
  for (int kk = 0; kk < K; kk += 32) {
    bvec16 av0, av1, bv0, bv1, bv2, bv3;
    av0.h2[0] = *(const bf16x8*)(arow0 + kk + kbA);
    av0.h2[1] = *(const bf16x8*)(arow0 + kk + kbA + 8);
    av1.h2[0] = *(const bf16x8*)(arow1 + kk + kbA);
    av1.h2[1] = *(const bf16x8*)(arow1 + kk + kbA + 8);
    bv0.h2[0] = *(const bf16x8*)(brow0 + kk + kbB);
    bv0.h2[1] = *(const bf16x8*)(brow0 + kk + kbB + 8);
    bv1.h2[0] = *(const bf16x8*)(brow1 + kk + kbB);
    bv1.h2[1] = *(const bf16x8*)(brow1 + kk + kbB + 8);
    bv2.h2[0] = *(const bf16x8*)(brow2 + kk + kbB);
    bv2.h2[1] = *(const bf16x8*)(brow2 + kk + kbB + 8);
    bv3.h2[0] = *(const bf16x8*)(brow3 + kk + kbB);
    bv3.h2[1] = *(const bf16x8*)(brow3 + kk + kbB + 8);
    if (kk + 128 < K) {
      __builtin_prefetch((const void*)(arow0 + kk + 128), 0, 1);
      __builtin_prefetch((const void*)(arow1 + kk + 128), 0, 1);
      __builtin_prefetch((const void*)(brow0 + kk + 128), 0, 1);
      __builtin_prefetch((const void*)(brow2 + kk + 128), 0, 1);
    }
    acc[0][0] = wmma_bf16(av0, bv0, acc[0][0]);
    acc[0][1] = wmma_bf16(av0, bv1, acc[0][1]);
    acc[0][2] = wmma_bf16(av0, bv2, acc[0][2]);
    acc[0][3] = wmma_bf16(av0, bv3, acc[0][3]);
    acc[1][0] = wmma_bf16(av1, bv0, acc[1][0]);
    acc[1][1] = wmma_bf16(av1, bv1, acc[1][1]);
    acc[1][2] = wmma_bf16(av1, bv2, acc[1][2]);
    acc[1][3] = wmma_bf16(av1, bv3, acc[1][3]);
  }

  const int moff = (lane < 16) ? 0 : 8;
#pragma unroll
  for (int i = 0; i < 2; ++i) {
#pragma unroll
    for (int j = 0; j < 4; ++j) {
      const int colb = tn * 64 + j * 16 + nloc;
      const float bval = bias ? bias[colb] : 0.0f;
#pragma unroll
      for (int r = 0; r < 8; ++r) {
        const int row = tm * 32 + i * 16 + r + moff;
        if (Cf) Cf[(size_t)row * N + colb] = acc[i][j][r] + bval;
        else    Cb[(size_t)row * N + colb] = f2bf(acc[i][j][r] + bval);
      }
    }
  }
}

// ---------------------------------------------------------------- attention
// One wave handles one (b, h, 16-row Q tile). Flash-style online softmax over
// 32-wide causal KV blocks; QK^T and P@V via v_wmma_f32_16x16x32_bf16.
// V blocks are staged to LDS with global_load_async_to_lds_b128, overlapped
// with the QK WMMAs and softmax of the same block.
__global__ __launch_bounds__(128) void k_attn_causal(
    const __bf16* __restrict__ qkv, float* __restrict__ yatt) {
  __shared__ __bf16 pbuf[4][16 * 32];    // per-wave P staging (C-layout -> A-layout)
  __shared__ __bf16 vstage[4][32 * 64];  // per-wave V block (row-major, 128B rows)

  const int lane = threadIdx.x & 31;
  const int wave = threadIdx.x >> 5;
  const int wid  = blockIdx.x * 4 + wave;
  const int qt   = wid & 127;          // T/16 tiles
  const int h    = (wid >> 7) & (NH - 1);
  const int b    = wid >> 11;
  const int qi   = qt * 16;

  const int nloc = lane & 15;
  const int kbA  = (lane < 16) ? 0 : 8;
  const int kbB  = (lane < 16) ? 0 : 8;
  const int moff = kbA;

  const __bf16* qbase = qkv + (size_t)(b * T_SEQ) * QKV_LD + h * DH;
  const __bf16* kbase = qbase + C_EMB;
  const __bf16* vbase = qbase + 2 * C_EMB;

  // LDS byte offset of this wave's V staging buffer (low 32 bits of the
  // generic LDS-aperture address are the workgroup-relative LDS offset).
  const unsigned vlds = (unsigned)(unsigned long long)(const void*)&vstage[wave][0]
                        + (unsigned)(lane * 128);
  const __bf16* vrow = vbase + (size_t)lane * QKV_LD;  // row s = s0 + lane

  // Q tile as WMMA A operand (16 x 64 -> two K-steps of 32)
  bvec16 qa[2];
  {
    const __bf16* qrow = qbase + (size_t)(qi + nloc) * QKV_LD;
    qa[0].h2[0] = *(const bf16x8*)(qrow + kbA);
    qa[0].h2[1] = *(const bf16x8*)(qrow + kbA + 16);
    qa[1].h2[0] = *(const bf16x8*)(qrow + 32 + kbA);
    qa[1].h2[1] = *(const bf16x8*)(qrow + 32 + kbA + 16);
  }

  f32x8 yacc[4] = {{}, {}, {}, {}};
  float Lr[8], Mr[8];
#pragma unroll
  for (int r = 0; r < 8; ++r) { Lr[r] = 0.0f; Mr[r] = -INFINITY; }

  const float scl = 0.125f;  // 1/sqrt(64)

  for (int s0 = 0; s0 <= qi; s0 += 32) {
    // ---- kick off async V-block staging (32 rows x 128B, lane l -> row s0+l)
    {
      const __bf16* g = vrow + (size_t)s0 * QKV_LD;
#pragma unroll
      for (int seg = 0; seg < 8; ++seg)
        async_ld_b128(vlds + seg * 16, g + seg * 8);
    }

    // ---- S = Q K^T (two 16x16 score tiles covering s_local 0..15 / 16..31)
    f32x8 sa = {}, sb = {};
    const __bf16* k0 = kbase + (size_t)(s0 + nloc) * QKV_LD;       // B-lane: K-row s0+n
    const __bf16* k1 = kbase + (size_t)(s0 + 16 + nloc) * QKV_LD;
#pragma unroll
    for (int kk = 0; kk < 2; ++kk) {
      bvec16 kv0, kv1;
      kv0.h2[0] = *(const bf16x8*)(k0 + kk * 32 + kbB);
      kv0.h2[1] = *(const bf16x8*)(k0 + kk * 32 + kbB + 16);
      kv1.h2[0] = *(const bf16x8*)(k1 + kk * 32 + kbB);
      kv1.h2[1] = *(const bf16x8*)(k1 + kk * 32 + kbB + 16);
      sa = wmma_bf16(qa[kk], kv0, sa);
      sb = wmma_bf16(qa[kk], kv1, sb);
    }

    // ---- scale + causal mask (lane covers col j, VGPR r covers row i)
#pragma unroll
    for (int r = 0; r < 8; ++r) {
      const int i = qi + r + moff;
      float a = sa[r] * scl;
      float c = sb[r] * scl;
      if (s0 + nloc > i)      a = -INFINITY;
      if (s0 + 16 + nloc > i) c = -INFINITY;
      sa[r] = a; sb[r] = c;
    }

    // ---- online softmax (row reductions across 16-lane N groups)
#pragma unroll
    for (int r = 0; r < 8; ++r) {
      float v = fmaxf(sa[r], sb[r]);
      v = fmaxf(v, __shfl_xor(v, 1));
      v = fmaxf(v, __shfl_xor(v, 2));
      v = fmaxf(v, __shfl_xor(v, 4));
      v = fmaxf(v, __shfl_xor(v, 8));
      const float mn = fmaxf(Mr[r], v);
      const float corr = __expf(Mr[r] - mn);
      Mr[r] = mn;
      const float p0 = __expf(sa[r] - mn);
      const float p1 = __expf(sb[r] - mn);
      sa[r] = p0; sb[r] = p1;
      float s = p0 + p1;
      s += __shfl_xor(s, 1);
      s += __shfl_xor(s, 2);
      s += __shfl_xor(s, 4);
      s += __shfl_xor(s, 8);
      Lr[r] = Lr[r] * corr + s;
      yacc[0][r] *= corr; yacc[1][r] *= corr;
      yacc[2][r] *= corr; yacc[3][r] *= corr;
    }

    // ---- P: C-layout -> LDS row-major -> A-layout
#pragma unroll
    for (int r = 0; r < 8; ++r) {
      const int m = r + moff;
      pbuf[wave][m * 32 + nloc]      = f2bf(sa[r]);
      pbuf[wave][m * 32 + 16 + nloc] = f2bf(sb[r]);
    }
    __asm volatile("s_wait_dscnt 0" ::: "memory");  // in-wave LDS RAW
    bvec16 pa;
#pragma unroll
    for (int e = 0; e < 16; ++e) {
      const int kidx = kbA + ((e < 8) ? e : e + 8);
      pa.e[e] = pbuf[wave][nloc * 32 + kidx];
    }

    // ---- Y += P @ V  (16x32 @ 32x64 -> four 16x16 WMMAs)
    // B-operand lane n = dd-local; elem e -> s_local = kbB + (e<8 ? e : e+8) (from LDS stage)
    wait_async0();
#pragma unroll
    for (int t2 = 0; t2 < 4; ++t2) {
      bvec16 bv;
#pragma unroll
      for (int e = 0; e < 16; ++e) {
        bv.e[e] = vstage[wave][(kbB + ((e < 8) ? e : e + 8)) * 64 + 16 * t2 + nloc];
      }
      yacc[t2] = wmma_bf16(pa, bv, yacc[t2]);
    }
  }

  // ---- normalize and store att@v (fp32, [B,T,C] with head-h column block)
#pragma unroll
  for (int t2 = 0; t2 < 4; ++t2) {
#pragma unroll
    for (int r = 0; r < 8; ++r) {
      const int row = qi + r + moff;
      const int col = h * DH + 16 * t2 + nloc;
      yatt[(size_t)(b * T_SEQ + row) * C_EMB + col] = yacc[t2][r] / Lr[r];
    }
  }
}

// ---------------------------------------------------------------- shaping
// y = beta*(att@v) + alpha*v - gamma*prefix_mean(v); one thread per (b,h,dd).
__global__ __launch_bounds__(256) void k_combine_shape(
    const float* __restrict__ yatt, const __bf16* __restrict__ qkv,
    __bf16* __restrict__ ybf,
    const float* __restrict__ alpha_p, const float* __restrict__ beta_p,
    const float* __restrict__ gamma_p) {
  const int idx = blockIdx.x * blockDim.x + threadIdx.x;  // 0..2047
  if (idx >= B_SZ * NH * DH) return;
  const int dd = idx & (DH - 1);
  const int h  = (idx >> 6) & (NH - 1);
  const int b  = idx >> 10;
  const float alpha = alpha_p[0], beta = beta_p[0], gamma = gamma_p[0];

  const __bf16* vcol = qkv + (size_t)(b * T_SEQ) * QKV_LD + 2 * C_EMB + h * DH + dd;
  float run = 0.0f;
  for (int t = 0; t < T_SEQ; ++t) {
    const float v = bf2f(vcol[(size_t)t * QKV_LD]);
    run += v;
    const size_t o = (size_t)(b * T_SEQ + t) * C_EMB + h * DH + dd;
    const float y = beta * yatt[o] + alpha * v - gamma * run / (float)(t + 1);
    ybf[o] = f2bf(y);
  }
}

// ---------------------------------------------------------------- launch
extern "C" void kernel_launch(void* const* d_in, const int* in_sizes, int n_in,
                              void* d_out, int out_size, void* d_ws, size_t ws_size,
                              hipStream_t stream) {
  (void)in_sizes; (void)n_in; (void)out_size; (void)ws_size;
  const float* x     = (const float*)d_in[0];
  const float* Wa    = (const float*)d_in[1];
  const float* ba    = (const float*)d_in[2];
  const float* Wp    = (const float*)d_in[3];
  const float* bp    = (const float*)d_in[4];
  const float* alpha = (const float*)d_in[5];
  const float* beta  = (const float*)d_in[6];
  const float* gamma = (const float*)d_in[7];

  const long NTOK = (long)B_SZ * T_SEQ;           // 4096
  size_t off = 0;
  auto wsp = [&](size_t bytes) {
    void* p = (char*)d_ws + off;
    off += (bytes + 255) & ~(size_t)255;
    return p;
  };
  __bf16* x_bf   = (__bf16*)wsp(NTOK * C_EMB * 2);           // 8.4 MB
  __bf16* wa_bf  = (__bf16*)wsp((size_t)QKV_LD * C_EMB * 2); // 6.3 MB
  __bf16* wp_bf  = (__bf16*)wsp((size_t)C_EMB * C_EMB * 2);  // 2.1 MB
  __bf16* qkv_bf = (__bf16*)wsp(NTOK * QKV_LD * 2);          // 25.2 MB
  float*  yatt   = (float*)wsp(NTOK * C_EMB * 4);            // 16.8 MB
  __bf16* y_bf   = (__bf16*)wsp(NTOK * C_EMB * 2);           // 8.4 MB

  // 1) fp32 -> bf16
  k_f32_to_bf16<<<2048, 256, 0, stream>>>(x,  x_bf,  NTOK * C_EMB);
  k_f32_to_bf16<<<1536, 256, 0, stream>>>(Wa, wa_bf, (long)QKV_LD * C_EMB);
  k_f32_to_bf16<<<512,  256, 0, stream>>>(Wp, wp_bf, (long)C_EMB * C_EMB);

  // 2) qkv = x @ Wa^T + ba   (M=4096, N=3072, K=1024) -> bf16
  {
    int tiles = (int)(NTOK / 32) * (QKV_LD / 64);  // 128 * 48 = 6144 waves
    k_gemm_bf16_wmma<<<tiles / 4, 128, 0, stream>>>(
        x_bf, wa_bf, ba, nullptr, qkv_bf, (int)NTOK, QKV_LD, C_EMB);
  }

  // 3) flash-style causal attention -> att@v (fp32)
  k_attn_causal<<<(B_SZ * NH * (T_SEQ / 16)) / 4, 128, 0, stream>>>(qkv_bf, yatt);

  // 4) shaped combine -> bf16 y
  k_combine_shape<<<(B_SZ * NH * DH + 255) / 256, 256, 0, stream>>>(
      yatt, qkv_bf, y_bf, alpha, beta, gamma);

  // 5) out = y @ Wp^T + bp   (M=4096, N=1024, K=1024) -> fp32 d_out
  {
    int tiles = (int)(NTOK / 32) * (C_EMB / 64);   // 128 * 16 = 2048 waves
    k_gemm_bf16_wmma<<<tiles / 4, 128, 0, stream>>>(
        y_bf, wp_bf, bp, (float*)d_out, nullptr, (int)NTOK, C_EMB, C_EMB);
  }
}
